// Rnn_35012573397619
// MI455X (gfx1250) — compile-verified, matrix operand order checked by reading, not run on hardware
//
#include <hip/hip_runtime.h>

// GRU (B=512, T=1024, I=64, H=128) + FC(51) for MI455X / gfx1250, wave32.
// f16 WMMA (16x16x32) with f32 accumulation. One block = 16 batch rows,
// 8 waves; wave j owns hidden columns [16j,16j+16). Weights live in
// registers as f16 B-fragments for the whole time loop. Hidden state and
// the x tile (double-buffered, converted to f16 once per step by all 256
// threads cooperatively) are staged through LDS. Gate activations use
// v_exp_f32 + v_rcp_f32 (TRANS32 ops co-issue with WMMA).

typedef __attribute__((ext_vector_type(16))) _Float16 v16h;
typedef __attribute__((ext_vector_type(8)))  _Float16 v8h;
typedef __attribute__((ext_vector_type(4)))  _Float16 v4h;
typedef __attribute__((ext_vector_type(8)))  float    v8f;
typedef __attribute__((ext_vector_type(4)))  float    v4f;

#define GRU_T 1024
#define GRU_I 64
#define GRU_H 128
#define GRU_O 51

#define WMMA_F16(A, B, C) \
  __builtin_amdgcn_wmma_f32_16x16x32_f16(false, (A), false, (B), (short)0, (C), false, false)

// Fast reciprocal: single v_rcp_f32 instead of IEEE div expansion.
__device__ __forceinline__ float frcp(float x) { return __builtin_amdgcn_rcpf(x); }

// A/B fragment fill per ISA 16-bit 16x32 layout:
// lane L holds row (L&15); element e<8 -> k = k0+e ; e>=8 -> k = k0+16+(e-8),
// where k0 = kbase + (L>>4)*8 is passed in by the caller.
__device__ __forceinline__ v16h frag_from_f32row(const float* __restrict__ row, int k0) {
  v16h r;
#pragma unroll
  for (int e = 0; e < 8; ++e) r[e] = (_Float16)row[k0 + e];
#pragma unroll
  for (int e = 0; e < 8; ++e) r[8 + e] = (_Float16)row[k0 + 16 + e];
  return r;
}

__device__ __forceinline__ v16h frag_from_ldsrow(const _Float16* row, int k0) {
  v8h lo = *(const v8h*)(row + k0);        // ds_load_b128
  v8h hi = *(const v8h*)(row + k0 + 16);   // ds_load_b128
  return __builtin_shufflevector(lo, hi, 0, 1, 2, 3, 4, 5, 6, 7,
                                         8, 9, 10, 11, 12, 13, 14, 15);
}

__device__ __forceinline__ v8f vsplat(float s) {
  v8f r;
#pragma unroll
  for (int e = 0; e < 8; ++e) r[e] = s;
  return r;
}

__device__ __forceinline__ v8f vsigmoid(v8f x) {
  v8f r;
#pragma unroll
  for (int e = 0; e < 8; ++e) r[e] = frcp(1.0f + __expf(-x[e]));
  return r;
}

__device__ __forceinline__ v8f vtanh(v8f x) {
  v8f r;
#pragma unroll
  for (int e = 0; e < 8; ++e) {
    float t = __expf(-2.0f * x[e]);
    r[e] = (1.0f - t) * frcp(1.0f + t);
  }
  return r;
}

__global__ __launch_bounds__(256) void gru_fused_kernel(
    const float* __restrict__ x,     // [B, 1, T, 64]
    const float* __restrict__ W_ih,  // [384, 64]
    const float* __restrict__ W_hh,  // [384, 128]
    const float* __restrict__ b_ih,  // [384]
    const float* __restrict__ b_hh,  // [384]
    const float* __restrict__ W_fc,  // [51, 128]
    const float* __restrict__ b_fc,  // [51]
    float* __restrict__ out)         // [B, 51]
{
  __shared__ alignas(16) _Float16 hbuf[16 * GRU_H];      // f16 hidden state, 16x128
  __shared__ alignas(16) _Float16 xbuf[2][16 * GRU_I];   // f16 x tile, double-buffered
  __shared__ float hfin[16 * GRU_H];                     // f32 final hidden for FC

  const int tid   = threadIdx.x;
  const int wave  = tid >> 5;          // 0..7 : hidden-column chunk
  const int lane  = tid & 31;
  const int lm    = lane & 15;         // row (A) / column (B) within a 16-tile
  const int lk8   = (lane >> 4) * 8;   // K sub-offset per fragment layout
  const int mhi   = (lane >> 4) * 8;   // high-M offset in C/D layout
  const int ncol  = wave * 16 + lm;    // hidden column this lane produces
  const int bbase = blockIdx.x * 16;   // batch-row base of this block

  // ---- Preload weight B-fragments (f32 -> f16) into registers ----
  v16h wih[3][2];  // gates r,z,n ; K-chunks of 32 over I=64
  v16h whh[3][4];  // gates r,z,n ; K-chunks of 32 over H=128
#pragma unroll
  for (int g = 0; g < 3; ++g) {
    const float* wih_row = W_ih + (size_t)(g * GRU_H + ncol) * GRU_I;
    const float* whh_row = W_hh + (size_t)(g * GRU_H + ncol) * GRU_H;
#pragma unroll
    for (int c = 0; c < 2; ++c) wih[g][c] = frag_from_f32row(wih_row, c * 32 + lk8);
#pragma unroll
    for (int c = 0; c < 4; ++c) whh[g][c] = frag_from_f32row(whh_row, c * 32 + lk8);
  }

  // Per-lane biases (constant over the row dimension of the tile)
  const float br  = b_ih[0 * GRU_H + ncol] + b_hh[0 * GRU_H + ncol];
  const float bz  = b_ih[1 * GRU_H + ncol] + b_hh[1 * GRU_H + ncol];
  const float bxn = b_ih[2 * GRU_H + ncol];
  const float bhn = b_hh[2 * GRU_H + ncol];

  // ---- x staging geometry: thread tid handles 4 consecutive floats ----
  const int srow = tid >> 4;           // 0..15 : batch row within tile
  const int scol = (tid & 15) * 4;     // 0..60 : input-feature offset
  const float* sx = x + (size_t)(bbase + srow) * (size_t)(GRU_T * GRU_I) + scol;

  // ---- Init hidden state; stage x[0] into buffer 0 ----
  v8f h_acc = vsplat(0.0f);
#pragma unroll
  for (int i = 0; i < 8; ++i) hbuf[tid * 8 + i] = (_Float16)0.0f;
  {
    v4f xv = *(const v4f*)sx;
    v4h xh;
#pragma unroll
    for (int e = 0; e < 4; ++e) xh[e] = (_Float16)xv[e];
    *(v4h*)(&xbuf[0][srow * GRU_I + scol]) = xh;
  }
  __syncthreads();

  // ---- Time loop ----
  for (int t = 0; t < GRU_T; ++t) {
    // Issue next timestep's global x load early; latency hides under WMMAs.
    v4f xnext = vsplat(0.0f).lo;  // dummy init
    if (t + 1 < GRU_T) {
      xnext = *(const v4f*)(sx + (size_t)(t + 1) * GRU_I);
      __builtin_prefetch(sx + (size_t)(t + 2) * GRU_I, 0, 0);  // global_prefetch_b8
    }

    v8f accr  = vsplat(br);
    v8f accz  = vsplat(bz);
    v8f accxn = vsplat(bxn);
    v8f acchn = vsplat(bhn);

    // x-side: K = 64 (2 chunks) into r, z, n-gates  (A from LDS f16)
    const _Float16* xrow = &xbuf[t & 1][lm * GRU_I];
#pragma unroll
    for (int c = 0; c < 2; ++c) {
      v16h a = frag_from_ldsrow(xrow, c * 32 + lk8);
      accr  = WMMA_F16(a, wih[0][c], accr);
      accz  = WMMA_F16(a, wih[1][c], accz);
      accxn = WMMA_F16(a, wih[2][c], accxn);
    }

    // h-side: K = 128 (4 chunks) into r, z, n-gates
#pragma unroll
    for (int c = 0; c < 4; ++c) {
      v16h a = frag_from_ldsrow(hbuf + lm * GRU_H, c * 32 + lk8);
      accr  = WMMA_F16(a, whh[0][c], accr);
      accz  = WMMA_F16(a, whh[1][c], accz);
      acchn = WMMA_F16(a, whh[2][c], acchn);
    }
    __syncthreads();  // all waves finished reading hbuf(t) / xbuf(t&1)

    v8f r = vsigmoid(accr);
    v8f z = vsigmoid(accz);
    v8f n = vtanh(accxn + r * acchn);
    h_acc = (vsplat(1.0f) - z) * n + z * h_acc;

    // scatter new h chunk to LDS (C/D layout: m = v + 8*(lane>>4), n = ncol)
#pragma unroll
    for (int v = 0; v < 8; ++v) hbuf[(v + mhi) * GRU_H + ncol] = (_Float16)h_acc[v];

    // stage x[t+1] (converted to f16) into the other buffer
    if (t + 1 < GRU_T) {
      v4h xh;
#pragma unroll
      for (int e = 0; e < 4; ++e) xh[e] = (_Float16)xnext[e];
      *(v4h*)(&xbuf[(t + 1) & 1][srow * GRU_I + scol]) = xh;
    }
    __syncthreads();  // hbuf(t+1), xbuf((t+1)&1) visible to all waves
  }

  // ---- FC head: out = hT @ W_fc^T + b_fc ----
#pragma unroll
  for (int v = 0; v < 8; ++v) hfin[(v + mhi) * GRU_H + ncol] = h_acc[v];
  __syncthreads();

  for (int idx = tid; idx < 16 * GRU_O; idx += 256) {
    const int m = idx / GRU_O;
    const int o = idx % GRU_O;
    float sum = b_fc[o];
    const float* wrow = W_fc + (size_t)o * GRU_H;
#pragma unroll 8
    for (int k = 0; k < GRU_H; ++k) sum += hfin[m * GRU_H + k] * wrow[k];
    out[(size_t)(bbase + m) * GRU_O + o] = sum;
  }
}

extern "C" void kernel_launch(void* const* d_in, const int* in_sizes, int n_in,
                              void* d_out, int out_size, void* d_ws, size_t ws_size,
                              hipStream_t stream) {
  const float* x    = (const float*)d_in[0];
  const float* W_ih = (const float*)d_in[1];
  const float* W_hh = (const float*)d_in[2];
  const float* b_ih = (const float*)d_in[3];
  const float* b_hh = (const float*)d_in[4];
  const float* W_fc = (const float*)d_in[5];
  const float* b_fc = (const float*)d_in[6];
  float* out = (float*)d_out;

  const int B = in_sizes[0] / (GRU_T * GRU_I);  // 512
  dim3 grid(B / 16), block(256);
  gru_fused_kernel<<<grid, block, 0, stream>>>(x, W_ih, W_hh, b_ih, b_hh, W_fc, b_fc, out);
}
